// ModConv2d_42142219108420
// MI455X (gfx1250) — matrix operand order repaired and checked by reading, not compiled, not run on hardware
//
#include <hip/hip_runtime.h>
#include <hip/hip_bf16.h>
#include <stdint.h>

// ---------------------------------------------------------------------------
// Modulated conv2d (StyleGAN2) for MI455X / gfx1250.
//  k_modstyle : s[b,c] = style . mod_weight^T * lin_scale + bias
//  k_wpack    : modulate + demodulate, pack bf16 weights, K-order k=tap*512+c
//  k_transpose: input f32 [B][C][64][64] -> bf16 channels-last padded
//               inT[B][66][66][C] (zero border = SAME padding for free)
//  k_conv     : per-sample implicit GEMM, double-buffered LDS staged with
//               async global->LDS b128 copies, v_wmma_f32_16x16x32_bf16 core
// ---------------------------------------------------------------------------

typedef __attribute__((ext_vector_type(16))) __bf16 bf16x16;
typedef __attribute__((ext_vector_type(8)))  __bf16 bf16x8;
typedef __attribute__((ext_vector_type(8)))  float  f32x8;
typedef __attribute__((ext_vector_type(4)))  int    i32x4;

#define BB   8
#define CC   512
#define OO   512
#define HH   64
#define WW   64
#define NPIX 4096
#define KTOT 4608          // C*9
#define HP   66
#define WP   66
#define LDSPITCH 40        // bf16 per LDS row (80B stride -> conflict-free cols)

#if defined(__HIP_DEVICE_COMPILE__) && \
    __has_builtin(__builtin_amdgcn_global_load_async_to_lds_b128) && \
    __has_builtin(__builtin_amdgcn_s_wait_asynccnt)
#define USE_ASYNC 1
#else
#define USE_ASYNC 0
#endif

typedef __attribute__((address_space(1))) i32x4 gas_i32x4;  // global int4
typedef __attribute__((address_space(3))) i32x4 las_i32x4;  // LDS int4

__device__ __forceinline__ void cp16(const void* g, void* l) {
#if USE_ASYNC
  __builtin_amdgcn_global_load_async_to_lds_b128(
      (gas_i32x4*)(uintptr_t)g, (las_i32x4*)(uint32_t)(uintptr_t)l, 0, 0);
#else
  *(uint4*)l = *(const uint4*)g;
#endif
}

__device__ __forceinline__ void wait_async() {
#if USE_ASYNC
  __builtin_amdgcn_s_wait_asynccnt(0);
#endif
}

__device__ __forceinline__ __bf16 f2bf(float f) {
  unsigned u = __builtin_bit_cast(unsigned, f);
  unsigned r = u + 0x7FFFu + ((u >> 16) & 1u);
  unsigned short h = (unsigned short)(r >> 16);
  return __builtin_bit_cast(__bf16, h);
}

// ---------------- Stage 1: EqualLinear modulation --------------------------
__global__ __launch_bounds__(256)
void k_modstyle(const float* __restrict__ style,
                const float* __restrict__ mod_weight,
                const float* __restrict__ mod_bias,
                float* __restrict__ Sbuf) {
  int gid = blockIdx.x * 256 + threadIdx.x;   // 4096 = 8*512
  int b = gid >> 9;
  int c = gid & 511;
  const float* st = style + b * 512;
  const float* mw = mod_weight + c * 512;
  float acc = 0.f;
#pragma unroll 4
  for (int j = 0; j < 512; ++j) acc += st[j] * mw[j];
  Sbuf[gid] = acc * 0.044194173824159216f + mod_bias[c];  // lin_scale=1/sqrt(512)
}

// ---------------- Stage 2: modulate + demodulate + pack bf16 ---------------
__global__ __launch_bounds__(256)
void k_wpack(const float* __restrict__ weight,   // [O][C][3][3]
             const float* __restrict__ Sbuf,     // [B][C]
             __bf16* __restrict__ Wmod) {        // [B][O][9*512], k=tap*512+c
  int bo = blockIdx.x;            // b*512 + o
  int b  = bo >> 9;
  int o  = bo & 511;
  int t  = threadIdx.x;

  const float* wrow = weight + (size_t)o * KTOT;  // flattened c*9+tap
  const float* srow = Sbuf + b * 512;

  float v[18];
  float ss = 0.f;
#pragma unroll
  for (int i = 0; i < 18; ++i) {
    int idx = t + i * 256;        // idx = c*9 + tap
    int c = idx / 9;
    float w = 0.014731391274719738f * wrow[idx] * srow[c];  // 1/sqrt(4608)
    v[i] = w;
    ss += w * w;
  }

  __shared__ float red[256];
  red[t] = ss;
  __syncthreads();
  for (int sft = 128; sft > 0; sft >>= 1) {
    if (t < sft) red[t] += red[t + sft];
    __syncthreads();
  }
  float demod = rsqrtf(red[0] + 1e-8f);

  __bf16* outp = Wmod + (size_t)bo * KTOT;
#pragma unroll
  for (int i = 0; i < 18; ++i) {
    int idx = t + i * 256;
    int c = idx / 9;
    int tap = idx - c * 9;
    outp[tap * 512 + c] = f2bf(v[i] * demod);
  }
}

// ---------------- Stage 2b: input -> channels-last padded bf16 -------------
// Grid (4 c-chunks, 64 y, 8 b), 256 threads. Border pre-zeroed by memset.
__global__ __launch_bounds__(256)
void k_transpose(const float* __restrict__ in,   // [B][C][64][64]
                 __bf16* __restrict__ inT) {     // [B][66][66][C]
  __shared__ __bf16 tile[128][66];
  const int b  = blockIdx.z;
  const int y  = blockIdx.y;
  const int c0 = blockIdx.x * 128;
  const int t  = threadIdx.x;

  {
    int cl = t >> 1;
    int xh = (t & 1) * 32;
    const float* src = in + (((size_t)b * CC + c0 + cl) * HH + y) * WW + xh;
#pragma unroll
    for (int i = 0; i < 32; ++i) tile[cl][xh + i] = f2bf(src[i]);
  }
  __syncthreads();
  {
    int xl = t >> 2;
    int cs = (t & 3) * 32;
    __bf16* dst = inT + (((size_t)(b * HP + y + 1)) * WP + xl + 1) * CC + c0 + cs;
#pragma unroll
    for (int i = 0; i < 32; i += 2) {
      unsigned lo = __builtin_bit_cast(unsigned short, tile[cs + i][xl]);
      unsigned hi = __builtin_bit_cast(unsigned short, tile[cs + i + 1][xl]);
      *(unsigned*)&dst[i] = lo | (hi << 16);
    }
  }
}

// ---------------- Stage 3: implicit-GEMM conv with WMMA --------------------
// Grid: (N/128 = 32, O/128 = 4, B = 8). Block: 256 threads = 8 waves (4Mx2N).
__global__ __launch_bounds__(256)
void k_conv(const __bf16* __restrict__ inT,     // [B][66][66][C]
            const __bf16* __restrict__ Wmod,    // [B][O][4608]
            float* __restrict__ out) {          // [B][O][64][64]
  __shared__ __bf16 Alds[2][128][LDSPITCH];     // 2 x 10 KB
  __shared__ __bf16 Blds[2][128][LDSPITCH];     // 2 x 10 KB

  const int t     = threadIdx.x;
  const int b     = blockIdx.z;
  const int Mbase = blockIdx.y * 128;
  const int nbase = blockIdx.x * 128;           // 128 pixels = 2 image rows

  const int lane = t & 31;
  const int wave = t >> 5;
  const int wm   = wave & 3;                    // M strip of 32
  const int wn   = wave >> 2;                   // N strip of 64
  const int half = lane >> 4;
  const int l16  = lane & 15;

  f32x8 acc[2][4] = {};

  // staging mapping: thread owns (row = t>>1, 16-elem segment = t&1), 32B each
  const int row = t >> 1;
  const int seg = (t & 1) * 16;
  const __bf16* gA0 = Wmod + (size_t)(b * OO + Mbase + row) * KTOT + seg;
  const int iy = (nbase + row) >> 6;
  const int ix = (nbase + row) & 63;
  const __bf16* gB0 = inT + ((size_t)(b * HP + iy) * WP + ix) * CC + seg;

  auto issue = [&](int step, int buf) {
    const int tap = step >> 4;                  // 0..8
    const int c0  = (step & 15) << 5;           // 0..480
    const int dy  = tap / 3;                    // padded-space offsets 0..2
    const int dx  = tap - dy * 3;
    const __bf16* gA = gA0 + tap * 512 + c0;
    const __bf16* gB = gB0 + ((size_t)dy * WP + dx) * CC + c0;
    __bf16* lA = &Alds[buf][row][seg];
    __bf16* lB = &Blds[buf][row][seg];
    cp16(gA, lA);
    cp16(gA + 8, lA + 8);
    cp16(gB, lB);
    cp16(gB + 8, lB + 8);
  };

  auto compute = [&](int buf) {
    // A 16x32 bf16: lane-half 0 -> K 0..7 & 16..23 ; half 1 -> K 8..15 & 24..31
    bf16x16 afrag[2];
#pragma unroll
    for (int mt = 0; mt < 2; ++mt) {
      const __bf16* ap = Alds[buf][wm * 32 + mt * 16 + l16];
      bf16x8 lo = *(const bf16x8*)(ap + 8 * half);
      bf16x8 hi = *(const bf16x8*)(ap + 16 + 8 * half);
      afrag[mt] = __builtin_shufflevector(lo, hi,
          0, 1, 2, 3, 4, 5, 6, 7, 8, 9, 10, 11, 12, 13, 14, 15);
    }
    // B 32x16 bf16: lane-half 0 -> K 0..7 & 16..23 ; half 1 -> K 8..15 & 24..31
#pragma unroll
    for (int nt = 0; nt < 4; ++nt) {
      const __bf16* bp = Blds[buf][wn * 64 + nt * 16 + l16] + 8 * half;
      bf16x8 blo = *(const bf16x8*)bp;
      bf16x8 bhi = *(const bf16x8*)(bp + 16);
      bf16x16 bfrag = __builtin_shufflevector(blo, bhi,
          0, 1, 2, 3, 4, 5, 6, 7, 8, 9, 10, 11, 12, 13, 14, 15);
#pragma unroll
      for (int mt = 0; mt < 2; ++mt) {
        acc[mt][nt] = __builtin_amdgcn_wmma_f32_16x16x32_bf16(
            false, afrag[mt], false, bfrag, (short)0, acc[mt][nt],
            false, false);
      }
    }
  };

  issue(0, 0);
  for (int step = 0; step < 144; ++step) {
    const int buf = step & 1;
    wait_async();          // own async copies for buf complete
    __syncthreads();       // everyone's copies landed; prior reads finished
    if (step + 1 < 144) issue(step + 1, buf ^ 1);  // overlap with compute
    compute(buf);
  }

  // epilogue: C/D layout -> VGPR r holds row (r + 8*half), col l16
  float* obase = out + (size_t)b * OO * NPIX;
#pragma unroll
  for (int mt = 0; mt < 2; ++mt) {
#pragma unroll
    for (int nt = 0; nt < 4; ++nt) {
      int n = nbase + wn * 64 + nt * 16 + l16;
#pragma unroll
      for (int r = 0; r < 8; ++r) {
        int m = Mbase + wm * 32 + mt * 16 + r + 8 * half;
        obase[(size_t)m * NPIX + n] = acc[mt][nt][r];
      }
    }
  }
}

// ---------------------------------------------------------------------------
extern "C" void kernel_launch(void* const* d_in, const int* in_sizes, int n_in,
                              void* d_out, int out_size, void* d_ws, size_t ws_size,
                              hipStream_t stream) {
  const float* input      = (const float*)d_in[0];  // [8,512,64,64]
  const float* style      = (const float*)d_in[1];  // [8,512]
  const float* weight     = (const float*)d_in[2];  // [1,512,512,3,3]
  const float* mod_weight = (const float*)d_in[3];  // [512,512]
  const float* mod_bias   = (const float*)d_in[4];  // [512]

  float*  Sbuf = (float*)d_ws;                                  // 16 KB
  __bf16* Wmod = (__bf16*)((char*)d_ws + 16384);                // 37,748,736 B
  __bf16* inT  = (__bf16*)((char*)d_ws + 16384 + 37748736);     // 35,684,352 B
  const size_t inT_bytes = (size_t)BB * HP * WP * CC * 2;

  (void)hipMemsetAsync(inT, 0, inT_bytes, stream);              // zero border
  k_modstyle<<<16, 256, 0, stream>>>(style, mod_weight, mod_bias, Sbuf);
  k_wpack<<<4096, 256, 0, stream>>>(weight, Sbuf, Wmod);
  k_transpose<<<dim3(4, 64, 8), 256, 0, stream>>>(input, inT);
  k_conv<<<dim3(32, 4, 8), 256, 0, stream>>>(inT, Wmod, (float*)d_out);
}
